// MambaBlock_50921132261641
// MI455X (gfx1250) — matrix operand order repaired and checked by reading, not compiled, not run on hardware
//
#include <hip/hip_runtime.h>
#include <hip/hip_fp16.h>
#include <cstdint>

// Mamba block for MI455X (gfx1250, wave32, WMMA).
// Big GEMMs on v_wmma_f32_16x16x32_f16 with 32x64 register tiles per wave
// (2 A-frags x 4 B-frags -> 8 WMMAs per k-step, ~0.75KB fragment traffic/WMMA).
// Everything else is streaming kernels; scan is register-resident per (b,d).

#define DM   1024           // D_MODEL
#define DI   2048           // D_INNER
#define DS   16             // D_STATE
#define TT   1024           // sequence length
#define NB   2              // batch
#define ROWS (NB * TT)      // 2048 token rows
#define NXZ  (2 * DI)       // 4096 (x_s | z)

typedef __attribute__((ext_vector_type(16))) _Float16 v16h;
typedef __attribute__((ext_vector_type(8)))  float    v8f;

__device__ __forceinline__ float silu_f(float x)     { return x / (1.0f + __expf(-x)); }
__device__ __forceinline__ float softplus_f(float x) { return (x > 20.0f) ? x : log1pf(__expf(x)); }

// ---------------------------------------------------------------- f32 -> f16
__global__ void cvt_f16_kernel(const float* __restrict__ in, __half* __restrict__ out, int n) {
  int i = blockIdx.x * blockDim.x + threadIdx.x;
  if (i < n) out[i] = __float2half(in[i]);
}

// ---------------------------------------------------------------- RMSNorm
__global__ void rmsnorm_kernel(const float* __restrict__ x, const float* __restrict__ g,
                               __half* __restrict__ xn) {
  __shared__ float red[256];
  const int row = blockIdx.x, tid = threadIdx.x;
  const float* xr = x + (size_t)row * DM;
  float ss = 0.0f;
  for (int c = tid; c < DM; c += 256) { float v = xr[c]; ss += v * v; }
  red[tid] = ss; __syncthreads();
  for (int k = 128; k > 0; k >>= 1) { if (tid < k) red[tid] += red[tid + k]; __syncthreads(); }
  const float scale = rsqrtf(red[0] * (1.0f / DM) + 1e-6f);
  for (int c = tid; c < DM; c += 256)
    xn[(size_t)row * DM + c] = __float2half(xr[c] * scale * g[c]);
}

// ---------------------------------------------------------------- WMMA GEMM
// Out[M,N] = (Cres ? Cres : 0) + A[M,K] * Bw[N,K]^T   (A,Bw f16 row-major)
// One wave computes a 32(M) x 64(N) register tile: c[2][4] accumulators.
// Fragment layouts per CDNA5 ISA 7.12.2:
//   A 16x32 f16: lane m = M%16, half h; per-lane bytes contiguous at
//                f16 offsets [8h..8h+7] and [16+8h..16+8h+7]  -> two b128
//   B 32x16    : lane n = N%16, half h; K = 8h+e and 16+8h+e, e=0..7     -> two b128
//   C/D 16x16  : VGPR r <-> M = r + 8h, N = lane%16
__global__ void wmma_gemm_kernel(const __half* __restrict__ Ah, const __half* __restrict__ Bh,
                                 const float* __restrict__ Cres, float* __restrict__ Out,
                                 int N, int K) {
  const _Float16* A  = reinterpret_cast<const _Float16*>(Ah);
  const _Float16* Bw = reinterpret_cast<const _Float16*>(Bh);
  const int lane = threadIdx.x & 31;
  const int hl   = lane >> 4;
  const int lm   = lane & 15;
  const int m0   = blockIdx.y * 32;
  const int n0   = blockIdx.x * 64;

  union U { v16h v; uint4 q[2]; };
  U a[2], b[4];
  v8f c[2][4];
#pragma unroll
  for (int i = 0; i < 2; ++i)
#pragma unroll
    for (int j = 0; j < 4; ++j)
#pragma unroll
      for (int r = 0; r < 8; ++r)
        c[i][j][r] = Cres ? Cres[(size_t)(m0 + 16 * i + r + 8 * hl) * N + n0 + 16 * j + lm]
                          : 0.0f;

  const _Float16* arow0 = A + (size_t)(m0 + lm) * K + 8 * hl;   // M rows m0+lm, m0+16+lm
  const _Float16* arow1 = arow0 + (size_t)16 * K;
  const _Float16* brow[4];
#pragma unroll
  for (int j = 0; j < 4; ++j)
    brow[j] = Bw + (size_t)(n0 + 16 * j + lm) * K + 8 * hl;

  for (int kk = 0; kk < K; kk += 32) {
    a[0].q[0] = *reinterpret_cast<const uint4*>(arow0 + kk);
    a[0].q[1] = *reinterpret_cast<const uint4*>(arow0 + kk + 16);
    a[1].q[0] = *reinterpret_cast<const uint4*>(arow1 + kk);
    a[1].q[1] = *reinterpret_cast<const uint4*>(arow1 + kk + 16);
#pragma unroll
    for (int j = 0; j < 4; ++j) {
      b[j].q[0] = *reinterpret_cast<const uint4*>(brow[j] + kk);
      b[j].q[1] = *reinterpret_cast<const uint4*>(brow[j] + kk + 16);
    }
#pragma unroll
    for (int i = 0; i < 2; ++i)
#pragma unroll
      for (int j = 0; j < 4; ++j)
        c[i][j] = __builtin_amdgcn_wmma_f32_16x16x32_f16(false, a[i].v, false, b[j].v,
                                                         (short)0, c[i][j], false, false);
  }

#pragma unroll
  for (int i = 0; i < 2; ++i)
#pragma unroll
    for (int j = 0; j < 4; ++j)
#pragma unroll
      for (int r = 0; r < 8; ++r)
        Out[(size_t)(m0 + 16 * i + r + 8 * hl) * N + n0 + 16 * j + lm] = c[i][j][r];
}

// ------------------------------------------- depthwise causal conv4 + SiLU
__global__ void conv_silu_kernel(const float* __restrict__ xz, const float* __restrict__ cw,
                                 const float* __restrict__ cb, float* __restrict__ xc) {
  int i = blockIdx.x * blockDim.x + threadIdx.x;
  if (i >= ROWS * DI) return;
  const int row = i / DI, d = i % DI;
  const int bb = row / TT, t = row % TT;
  float acc = cb[d];
#pragma unroll
  for (int j = 0; j < 4; ++j) {
    const int tt = t - 3 + j;
    if (tt >= 0) acc += cw[d * 4 + j] * xz[(size_t)(bb * TT + tt) * NXZ + d];
  }
  xc[i] = silu_f(acc);
}

// ------------------------------------------- params = xc @ W_x^T  (33 outs)
__global__ void params_kernel(const float* __restrict__ xc, const float* __restrict__ Wx,
                              float* __restrict__ par) {
  __shared__ float red[256];
  const int j = blockIdx.x, row = blockIdx.y, tid = threadIdx.x;
  const float* xr = xc + (size_t)row * DI;
  const float* wr = Wx + (size_t)j * DI;
  float s = 0.0f;
  for (int d = tid; d < DI; d += 256) s += xr[d] * wr[d];
  red[tid] = s; __syncthreads();
  for (int k = 128; k > 0; k >>= 1) { if (tid < k) red[tid] += red[tid + k]; __syncthreads(); }
  if (tid == 0) par[(size_t)row * 33 + j] = red[0];
}

// ------------------------------------------- selective scan: 1 thread/(b,d)
__global__ void scan_kernel(const float* __restrict__ par, const float* __restrict__ xc,
                            const float* __restrict__ A_log, const float* __restrict__ W_dt,
                            const float* __restrict__ b_dt, float* __restrict__ ybuf) {
  const int tid = blockIdx.x * blockDim.x + threadIdx.x;
  const int bb = tid / DI, d = tid % DI;
  float An[DS], h[DS];
#pragma unroll
  for (int s = 0; s < DS; ++s) { An[s] = -__expf(A_log[d * DS + s]); h[s] = 0.0f; }
  const float wdt = W_dt[d];
  const float bdt = b_dt[d];
  for (int t = 0; t < TT; ++t) {
    const int row = bb * TT + t;
    const float* p = par + (size_t)row * 33;
    const float dtv = softplus_f(p[32] * wdt + bdt);
    const float dx  = dtv * xc[(size_t)row * DI + d];
    float y = 0.0f;
#pragma unroll
    for (int s = 0; s < DS; ++s) {
      const float ab = __expf(dtv * An[s]);
      h[s] = ab * h[s] + dx * p[s];        // B_coef at p[0..15]
      y   += h[s] * p[16 + s];             // C_coef at p[16..31]
    }
    ybuf[(size_t)row * DI + d] = y;
  }
}

// ------------------------------------------- (y + xc*D) * silu(z) -> f16
__global__ void combine_kernel(const float* __restrict__ ybuf, const float* __restrict__ xc,
                               const float* __restrict__ Dp, const float* __restrict__ xz,
                               __half* __restrict__ yh) {
  int i = blockIdx.x * blockDim.x + threadIdx.x;
  if (i >= ROWS * DI) return;
  const int row = i / DI, d = i % DI;
  const float z = xz[(size_t)row * NXZ + DI + d];
  yh[i] = __float2half((ybuf[i] + xc[i] * Dp[d]) * silu_f(z));
}

// ============================================================ host launcher
extern "C" void kernel_launch(void* const* d_in, const int* in_sizes, int n_in,
                              void* d_out, int out_size, void* d_ws, size_t ws_size,
                              hipStream_t stream) {
  const float* x      = (const float*)d_in[0];
  const float* g      = (const float*)d_in[1];
  const float* W_in   = (const float*)d_in[2];
  const float* W_out  = (const float*)d_in[3];
  const float* conv_w = (const float*)d_in[4];
  const float* conv_b = (const float*)d_in[5];
  const float* W_x    = (const float*)d_in[6];
  const float* W_dt   = (const float*)d_in[7];
  const float* b_dt   = (const float*)d_in[8];
  const float* A_log  = (const float*)d_in[9];
  const float* Dp     = (const float*)d_in[10];

  char* w = (char*)d_ws;
  size_t off = 0;
  auto take = [&](size_t bytes) -> char* {
    char* p = w + off; off += (bytes + 255) & ~(size_t)255; return p;
  };
  __half* xn_h   = (__half*)take((size_t)ROWS * DM * 2);
  __half* win_h  = (__half*)take((size_t)NXZ  * DM * 2);
  __half* wout_h = (__half*)take((size_t)DM   * DI * 2);
  float*  xz     = (float*) take((size_t)ROWS * NXZ * 4);
  float*  xc     = (float*) take((size_t)ROWS * DI  * 4);
  float*  par    = (float*) take((size_t)ROWS * 33  * 4);
  float*  ybuf   = (float*) take((size_t)ROWS * DI  * 4);
  __half* yh     = (__half*)take((size_t)ROWS * DI  * 2);

  { int n = NXZ * DM; cvt_f16_kernel<<<(n + 255) / 256, 256, 0, stream>>>(W_in,  win_h,  n); }
  { int n = DM  * DI; cvt_f16_kernel<<<(n + 255) / 256, 256, 0, stream>>>(W_out, wout_h, n); }

  rmsnorm_kernel<<<ROWS, 256, 0, stream>>>(x, g, xn_h);

  // xz[2048,4096] = xn[2048,1024] @ W_in^T   (K = 1024), 32x64 tile per wave
  wmma_gemm_kernel<<<dim3(NXZ / 64, ROWS / 32), 32, 0, stream>>>(xn_h, win_h, nullptr, xz, NXZ, DM);

  { int n = ROWS * DI; conv_silu_kernel<<<(n + 255) / 256, 256, 0, stream>>>(xz, conv_w, conv_b, xc); }

  params_kernel<<<dim3(33, ROWS), 256, 0, stream>>>(xc, W_x, par);

  scan_kernel<<<(NB * DI) / 256, 256, 0, stream>>>(par, xc, A_log, W_dt, b_dt, ybuf);

  { int n = ROWS * DI; combine_kernel<<<(n + 255) / 256, 256, 0, stream>>>(ybuf, xc, Dp, xz, yh); }

  // out[2048,1024] = x + y2[2048,2048] @ W_out^T   (K = 2048, residual fused as C)
  wmma_gemm_kernel<<<dim3(DM / 64, ROWS / 32), 32, 0, stream>>>(yh, wout_h, x, (float*)d_out, DM, DI);
}
